// TileEarlyJoinGConv_13228499272259
// MI455X (gfx1250) — matrix operand order repaired and checked by reading, not compiled, not run on hardware
//
#include <hip/hip_runtime.h>
#include <hip/hip_bf16.h>
#include <stdint.h>

// ---------------------------------------------------------------------------
// Model constants (from the reference)
// ---------------------------------------------------------------------------
#define N_NODES   400
#define NB        10
#define N_CFG     1000
#define CFG_DIM   24
#define NODE_DIM  85
#define OP_EMB_D  32
#define SHAPE_D   4
#define HD        64
#define NE        800
#define MROWS     (N_NODES * N_CFG)      // 400,000 (node-major rows)
#define NODE_COLS 86                     // node_feat has 85 cont + shape col
#define ROWT      4                      // 16-row tiles per block (64 rows)

typedef float  v8f   __attribute__((ext_vector_type(8)));
typedef __bf16 v16bf __attribute__((ext_vector_type(16)));
typedef unsigned u32x4 __attribute__((ext_vector_type(4)));
typedef unsigned tdm_u4 __attribute__((ext_vector_type(4)));
typedef int      tdm_i8 __attribute__((ext_vector_type(8)));
typedef int      tdm_i4 __attribute__((ext_vector_type(4)));

// ---------------------------------------------------------------------------
// bf16 helpers (round-to-nearest-even)
// ---------------------------------------------------------------------------
__device__ __forceinline__ unsigned short f2bf(float f) {
    unsigned u = __builtin_bit_cast(unsigned, f);
    u += 0x7FFFu + ((u >> 16) & 1u);
    return (unsigned short)(u >> 16);
}
__device__ __forceinline__ float bf2f(unsigned short h) {
    unsigned u = ((unsigned)h) << 16;
    return __builtin_bit_cast(float, u);
}

// ---------------------------------------------------------------------------
// Tensor Data Mover: stage a [rows x K] bf16 row-major panel into LDS.
// D# fields per CDNA5 ISA (2D tile, groups 2/3 = 0):
//   group0: count=1 | lds_addr | global_addr(57b) | type=2
//   group1: data_size=1(2B), tensor_dim0=K, tensor_dim1=MROWS,
//           tile_dim0=K, tile_dim1=rows, tensor_dim0_stride=K
// All inputs are uniform (kernarg/blockIdx) so the descriptor stays in SGPRs.
// This toolchain exposes the 6-arg builtin (clang-23 / therock-10.0 form).
// ---------------------------------------------------------------------------
__device__ __forceinline__ void tdm_load_panel(const void* gptr, unsigned lds_off,
                                               int K, int rows) {
    uint64_t ga = (uint64_t)(uintptr_t)gptr;
    tdm_u4 g0;
    g0[0] = 1u;                                        // count=1 (valid user D#)
    g0[1] = lds_off;                                   // LDS byte address
    g0[2] = (unsigned)(ga & 0xFFFFFFFFu);              // global_addr[31:0]
    g0[3] = (unsigned)((ga >> 32) & 0x01FFFFFFu) | (2u << 30);   // addr hi | type=2
    tdm_i8 g1;
    g1[0] = 0x00010000;                                // data_size=1 -> 2 bytes/elem
    g1[1] = (K & 0xFFFF) << 16;                        // tensor_dim0[15:0]
    g1[2] = ((unsigned)K >> 16) | ((MROWS & 0xFFFF) << 16);      // td0 hi | td1 lo
    g1[3] = (MROWS >> 16) | ((K & 0xFFFF) << 16);      // td1 hi | tile_dim0=K
    g1[4] = rows & 0xFFFF;                             // tile_dim1=rows, tile_dim2=0
    g1[5] = K;                                         // tensor_dim0_stride[31:0]
    g1[6] = 0;                                         // stride hi, dim1_stride lo
    g1[7] = 0;
    tdm_i4 gz4 = {0, 0, 0, 0};
    tdm_i8 gz8 = {0, 0, 0, 0, 0, 0, 0, 0};
    __builtin_amdgcn_tensor_load_to_lds(g0, g1, gz4, gz4, gz8, 0);
}

// ---------------------------------------------------------------------------
// Weight packing: float [K][N] row-major -> dword-packed bf16 K-pairs
// out[kp*N + n] = { bf16(W[2kp][n]) , bf16(W[2kp+1][n]) }  (lo = even K)
// Matches the WMMA 16-bit B-fragment layout (kpair = h*4+v(+4 for v>=4), col = lane&15).
// ---------------------------------------------------------------------------
__global__ void pack_pairs_k(const float* __restrict__ W, unsigned* __restrict__ out,
                             int K, int N) {
    int idx = blockIdx.x * blockDim.x + threadIdx.x;
    int total = (K >> 1) * N;
    if (idx >= total) return;
    int kp = idx / N, n = idx % N;
    unsigned short lo = f2bf(W[(2 * kp) * N + n]);
    unsigned short hi = f2bf(W[(2 * kp + 1) * N + n]);
    out[idx] = (unsigned)lo | ((unsigned)hi << 16);
}

// ---------------------------------------------------------------------------
// Early-join static parts:  a[node][128] = b1 + xn(node) @ W1[:121]
// ---------------------------------------------------------------------------
__global__ void node_static_k(const float* __restrict__ node_feat,
                              const int*   __restrict__ opcode,
                              const float* __restrict__ op_emb,
                              const float* __restrict__ shape_emb,
                              const float* __restrict__ W1,
                              const float* __restrict__ b1,
                              float* __restrict__ a) {
    int i = blockIdx.x;          // node
    int j = threadIdx.x;         // 0..127
    const float* nf = node_feat + (size_t)i * NODE_COLS;
    float s = b1[j];
    for (int k = 0; k < NODE_DIM; ++k) s += nf[k] * W1[k * 128 + j];
    int st = (int)nf[NODE_DIM];
    for (int k = 0; k < SHAPE_D; ++k) s += shape_emb[st * SHAPE_D + k] * W1[(NODE_DIM + k) * 128 + j];
    int op = opcode[i];
    for (int k = 0; k < OP_EMB_D; ++k) s += op_emb[op * OP_EMB_D + k] * W1[(NODE_DIM + SHAPE_D + k) * 128 + j];
    a[(size_t)i * 128 + j] = s;
}

// C[gc][128] = cfg(gc) @ W1[121:145]
__global__ void cfg_static_k(const float* __restrict__ config_feat,
                             const float* __restrict__ W1,
                             float* __restrict__ C) {
    int gc = blockIdx.x;         // 0..9999
    int j  = threadIdx.x;        // 0..127
    const float* cf = config_feat + (size_t)gc * CFG_DIM;
    float s = 0.f;
    for (int k = 0; k < CFG_DIM; ++k) s += cf[k] * W1[(121 + k) * 128 + j];
    C[(size_t)gc * 128 + j] = s;
}

// x1[node][cfg][128] = relu(a[node] + C[batch[node]*1000+cfg])  -> bf16 dwords
__global__ void build_x1_k(const float* __restrict__ a, const float* __restrict__ C,
                           const int* __restrict__ batch, unsigned* __restrict__ x1) {
    size_t idx = (size_t)blockIdx.x * blockDim.x + threadIdx.x; // dword index, 25.6M
    int r  = (int)(idx >> 6);         // row = node*1000+cfg
    int jp = (int)(idx & 63);         // dword within row (2 feats)
    int node = r / N_CFG;
    int c    = r % N_CFG;
    int g    = batch[node];
    size_t ai = (size_t)node * 128 + 2 * jp;
    size_t ci = ((size_t)g * N_CFG + c) * 128 + 2 * jp;
    float v0 = fmaxf(a[ai]     + C[ci],     0.f);
    float v1 = fmaxf(a[ai + 1] + C[ci + 1], 0.f);
    x1[idx] = (unsigned)f2bf(v0) | ((unsigned)f2bf(v1) << 16);
}

// ---------------------------------------------------------------------------
// WMMA fragment loaders (v_wmma_f32_16x16x32_bf16), ISA §7.12.2 layouts:
//   A 16x32 bf16: lane = h*16+m holds row m; dwords {k0/2+h*4..+3, k0/2+8+h*4..+3}
//   B 32x16 bf16: lane col = m; dwords at kpair = k0/2 + h*4 + v (+4 for v>=4), stride N
// ---------------------------------------------------------------------------
union FragA { v16bf v; u32x4 q[2]; };
union FragB { v16bf v; unsigned u[8]; };

__device__ __forceinline__ v16bf load_fragA(const unsigned* xrow, int k0, int h) {
    FragA A;
    int s1 = (k0 >> 1) + h * 4;
    A.q[0] = *(const u32x4*)(xrow + s1);
    A.q[1] = *(const u32x4*)(xrow + s1 + 8);
    return A.v;
}
__device__ __forceinline__ v16bf load_fragB(const unsigned* __restrict__ Wp,
                                            int k0, int h, int m, int colBase, int N) {
    FragB B;
    int kb = (k0 >> 1) + h * 4;
#pragma unroll
    for (int v = 0; v < 8; ++v) B.u[v] = Wp[(size_t)(kb + v + (v >= 4 ? 4 : 0)) * N + colBase + m];
    return B.v;
}

// ---------------------------------------------------------------------------
// Y = relu(X @ W + b): X MxK bf16 row-major, W packed, Y MxN bf16.
// One block = 64-row panel staged via TDM into LDS; N/16 waves = col tiles;
// B fragments hoisted into registers and reused over 4 row tiles.
// K, N compile-time so the K-loop is straight-line ds_load->wmma code.
// ---------------------------------------------------------------------------
template <int K, int N>
__global__ void gemm_bias_relu_k(const unsigned* __restrict__ X,
                                 const unsigned* __restrict__ Wp,
                                 const float* __restrict__ b,
                                 unsigned short* __restrict__ Y) {
    constexpr int NK = K / 32;
    __shared__ alignas(16) unsigned short Ap[64 * K];
    int tid  = threadIdx.x;
    int wave = tid >> 5, lane = tid & 31;
    int h = lane >> 4, m = lane & 15;
    int rowPanel = blockIdx.x * (ROWT * 16);
    int colBase  = wave * 16;

    if (tid < 32) {   // wave 0 drives the Tensor Data Mover
        tdm_load_panel((const char*)X + (size_t)rowPanel * K * 2,
                       (unsigned)(uintptr_t)&Ap[0], K, ROWT * 16);
        __builtin_amdgcn_s_wait_tensorcnt(0);
    }
    __syncthreads();

    v16bf Bfr[NK];
#pragma unroll
    for (int kk = 0; kk < NK; ++kk)
        Bfr[kk] = load_fragB(Wp, kk * 32, h, m, colBase, N);

    float bias = b[colBase + m];
#pragma unroll
    for (int rt = 0; rt < ROWT; ++rt) {
        const unsigned* arow = (const unsigned*)&Ap[0] + (size_t)(rt * 16 + m) * (K >> 1);
        v8f c = {0.f, 0.f, 0.f, 0.f, 0.f, 0.f, 0.f, 0.f};
#pragma unroll
        for (int kk = 0; kk < NK; ++kk) {
            v16bf Av = load_fragA(arow, kk * 32, h);    // ds_load_b128 x2
            c = __builtin_amdgcn_wmma_f32_16x16x32_bf16(false, Av, false, Bfr[kk],
                                                        (short)0, c, false, false);
        }
#pragma unroll
        for (int j = 0; j < 8; ++j) {
            float val = fmaxf(c[j] + bias, 0.f);
            Y[(size_t)(rowPanel + rt * 16 + j + h * 8) * N + colBase + m] = f2bf(val);
        }
    }
}

// Y = relu(X @ Wl + Mn @ Wr + b), N = 64; both panels staged via TDM.
template <int K>
__global__ void sage_bias_relu_k(const unsigned* __restrict__ X,
                                 const unsigned* __restrict__ Mn,
                                 const unsigned* __restrict__ Wlp,
                                 const unsigned* __restrict__ Wrp,
                                 const float* __restrict__ b,
                                 unsigned short* __restrict__ Y) {
    constexpr int N = HD;
    constexpr int NK = K / 32;
    __shared__ alignas(16) unsigned short Xp[64 * K];
    __shared__ alignas(16) unsigned short Mp[64 * K];
    int tid  = threadIdx.x;
    int wave = tid >> 5, lane = tid & 31;
    int h = lane >> 4, m = lane & 15;
    int rowPanel = blockIdx.x * (ROWT * 16);
    int colBase  = wave * 16;

    if (tid < 32) {
        tdm_load_panel((const char*)X  + (size_t)rowPanel * K * 2,
                       (unsigned)(uintptr_t)&Xp[0], K, ROWT * 16);
        tdm_load_panel((const char*)Mn + (size_t)rowPanel * K * 2,
                       (unsigned)(uintptr_t)&Mp[0], K, ROWT * 16);
        __builtin_amdgcn_s_wait_tensorcnt(0);
    }
    __syncthreads();

    v16bf Bl[NK], Br[NK];
#pragma unroll
    for (int kk = 0; kk < NK; ++kk) {
        Bl[kk] = load_fragB(Wlp, kk * 32, h, m, colBase, N);
        Br[kk] = load_fragB(Wrp, kk * 32, h, m, colBase, N);
    }

    float bias = b[colBase + m];
#pragma unroll
    for (int rt = 0; rt < ROWT; ++rt) {
        const unsigned* xrow = (const unsigned*)&Xp[0] + (size_t)(rt * 16 + m) * (K >> 1);
        const unsigned* mrow = (const unsigned*)&Mp[0] + (size_t)(rt * 16 + m) * (K >> 1);
        v8f c = {0.f, 0.f, 0.f, 0.f, 0.f, 0.f, 0.f, 0.f};
#pragma unroll
        for (int kk = 0; kk < NK; ++kk) {
            v16bf Av = load_fragA(xrow, kk * 32, h);
            c = __builtin_amdgcn_wmma_f32_16x16x32_bf16(false, Av, false, Bl[kk],
                                                        (short)0, c, false, false);
        }
#pragma unroll
        for (int kk = 0; kk < NK; ++kk) {
            v16bf Av = load_fragA(mrow, kk * 32, h);
            c = __builtin_amdgcn_wmma_f32_16x16x32_bf16(false, Av, false, Br[kk],
                                                        (short)0, c, false, false);
        }
#pragma unroll
        for (int j = 0; j < 8; ++j) {
            float val = fmaxf(c[j] + bias, 0.f);
            Y[(size_t)(rowPanel + rt * 16 + j + h * 8) * N + colBase + m] = f2bf(val);
        }
    }
}

// ---------------------------------------------------------------------------
// Graph aggregation
// ---------------------------------------------------------------------------
__global__ void deg_k(const int* __restrict__ ei, float* __restrict__ deg) {
    int e = blockIdx.x * blockDim.x + threadIdx.x;
    if (e >= NE) return;
    atomicAdd(&deg[ei[NE + e]], 1.0f);
}

// agg[dst][cfg][f] += x[src][cfg][f] per edge; shift = log2(feat)
__global__ void edge_scatter_k(const unsigned short* __restrict__ X,
                               float* __restrict__ agg,
                               const int* __restrict__ ei, int shift) {
    int e = blockIdx.x;
    int t = blockIdx.y * blockDim.x + threadIdx.x;
    int feat = 1 << shift;
    if (t >= N_CFG * feat) return;
    int c = t >> shift;
    int f = t & (feat - 1);
    int src = ei[e], dst = ei[NE + e];
    float v = bf2f(X[((size_t)src * N_CFG + c) * feat + f]);
    atomicAdd(&agg[((size_t)dst * N_CFG + c) * feat + f], v);
}

// mean = agg / max(deg,1) -> bf16
__global__ void mean_fin_k(const float* __restrict__ agg,
                           const float* __restrict__ deg,
                           unsigned short* __restrict__ mean, int shift) {
    size_t idx = (size_t)blockIdx.x * blockDim.x + threadIdx.x;
    int r = (int)(idx >> shift);           // row = node*1000+cfg
    int node = r / N_CFG;
    float d = fmaxf(deg[node], 1.0f);
    mean[idx] = f2bf(agg[idx] / d);
}

// ---------------------------------------------------------------------------
// Pool (max + mean over nodes) + L2 normalize + MLP head, one block per (g,c)
// ---------------------------------------------------------------------------
__global__ void pool_head_k(const unsigned short* __restrict__ X3,
                            const float* __restrict__ Wp1, const float* __restrict__ bp1,
                            const float* __restrict__ Wp2, const float* __restrict__ bp2,
                            float* __restrict__ out) {
    int g = blockIdx.x / N_CFG;
    int c = blockIdx.x % N_CFG;
    int f = threadIdx.x;                    // 0..63
    const int npg = N_NODES / NB;           // 40 nodes per graph

    float mx = -__builtin_inff();
    float sm = 0.f;
    for (int n = g * npg; n < (g + 1) * npg; ++n) {
        if (n + 1 < (g + 1) * npg)
            __builtin_prefetch(&X3[((size_t)(n + 1) * N_CFG + c) * HD + f], 0, 0);
        float v = bf2f(X3[((size_t)n * N_CFG + c) * HD + f]);
        mx = fmaxf(mx, v);
        sm += v;
    }
    float xg = mx + sm / (float)npg;

    __shared__ float sh[HD];
    __shared__ float xs[HD];
    __shared__ float hb[HD / 2];
    sh[f] = xg * xg;
    __syncthreads();
    for (int s = HD / 2; s > 0; s >>= 1) {
        if (f < s) sh[f] += sh[f + s];
        __syncthreads();
    }
    float inv = 1.0f / sqrtf(sh[0]);
    xs[f] = xg * inv;
    __syncthreads();
    if (f < 32) {
        float h = bp1[f];
        for (int k = 0; k < HD; ++k) h += xs[k] * Wp1[k * 32 + f];
        hb[f] = fmaxf(h, 0.f);
    }
    __syncthreads();
    if (f == 0) {
        float o = bp2[0];
        for (int j = 0; j < 32; ++j) o += hb[j] * Wp2[j];
        out[blockIdx.x] = o;
    }
}

// ---------------------------------------------------------------------------
// Host orchestration
// ---------------------------------------------------------------------------
static inline size_t alup(size_t x) { return (x + 255) & ~(size_t)255; }

extern "C" void kernel_launch(void* const* d_in, const int* in_sizes, int n_in,
                              void* d_out, int out_size, void* d_ws, size_t ws_size,
                              hipStream_t stream) {
    (void)in_sizes; (void)n_in; (void)out_size; (void)ws_size;

    const float* node_feat   = (const float*)d_in[0];
    const int*   node_opcode = (const int*)  d_in[1];
    const float* config_feat = (const float*)d_in[2];
    const int*   edge_index  = (const int*)  d_in[3];
    const int*   batch       = (const int*)  d_in[4];
    const float* op_emb      = (const float*)d_in[5];
    const float* shape_emb   = (const float*)d_in[6];
    const float* W1  = (const float*)d_in[7];
    const float* b1  = (const float*)d_in[8];
    const float* W2  = (const float*)d_in[9];
    const float* b2  = (const float*)d_in[10];
    const float* Wl0 = (const float*)d_in[11];
    const float* Wr0 = (const float*)d_in[12];
    const float* bg0 = (const float*)d_in[13];
    const float* Wl1 = (const float*)d_in[14];
    const float* Wr1 = (const float*)d_in[15];
    const float* bg1 = (const float*)d_in[16];
    const float* Wl2 = (const float*)d_in[17];
    const float* Wr2 = (const float*)d_in[18];
    const float* bg2 = (const float*)d_in[19];
    const float* Wp1 = (const float*)d_in[20];
    const float* bp1 = (const float*)d_in[21];
    const float* Wp2 = (const float*)d_in[22];
    const float* bp2 = (const float*)d_in[23];
    float* out = (float*)d_out;

    // ---- workspace carve-up ----
    char* ws = (char*)d_ws;
    size_t o = 0;
    unsigned* W2p  = (unsigned*)(ws + o); o = alup(o + (size_t)64 * 128 * 4);
    unsigned* Wl0p = (unsigned*)(ws + o); o = alup(o + (size_t)64 * 64 * 4);
    unsigned* Wr0p = (unsigned*)(ws + o); o = alup(o + (size_t)64 * 64 * 4);
    unsigned* Wl1p = (unsigned*)(ws + o); o = alup(o + (size_t)32 * 64 * 4);
    unsigned* Wr1p = (unsigned*)(ws + o); o = alup(o + (size_t)32 * 64 * 4);
    unsigned* Wl2p = (unsigned*)(ws + o); o = alup(o + (size_t)32 * 64 * 4);
    unsigned* Wr2p = (unsigned*)(ws + o); o = alup(o + (size_t)32 * 64 * 4);
    float* a_node  = (float*)(ws + o);    o = alup(o + (size_t)N_NODES * 128 * 4);
    float* Cst     = (float*)(ws + o);    o = alup(o + (size_t)NB * N_CFG * 128 * 4);
    float* deg     = (float*)(ws + o);    o = alup(o + (size_t)N_NODES * 4);
    char* x1_region = ws + o;             o = alup(o + (size_t)MROWS * 128 * 2);  // 102.4 MB
    char* x2_region = ws + o;             o = alup(o + (size_t)MROWS * 128 * 2);  // 102.4 MB
    unsigned short* meanb = (unsigned short*)(ws + o); o = alup(o + (size_t)MROWS * 128 * 2);
    float* agg = (float*)(ws + o);        o = alup(o + (size_t)MROWS * 128 * 4);  // 204.8 MB

    unsigned*       x1   = (unsigned*)x1_region;              // bf16 dwords, K=128
    unsigned short* x2   = (unsigned short*)x2_region;        // K=128 activations
    unsigned short* x3a  = (unsigned short*)x1_region;        // reuse x1 (dead)
    unsigned short* x3b  = (unsigned short*)(x1_region + (size_t)MROWS * HD * 2);
    unsigned short* x3c  = (unsigned short*)x2_region;        // reuse x2 (dead)

    // ---- 1. pack weights to WMMA-B bf16 pair layout ----
    pack_pairs_k<<<32, 256, 0, stream>>>(W2,  W2p,  128, 128);
    pack_pairs_k<<<16, 256, 0, stream>>>(Wl0, Wl0p, 128, 64);
    pack_pairs_k<<<16, 256, 0, stream>>>(Wr0, Wr0p, 128, 64);
    pack_pairs_k<<<8,  256, 0, stream>>>(Wl1, Wl1p, 64, 64);
    pack_pairs_k<<<8,  256, 0, stream>>>(Wr1, Wr1p, 64, 64);
    pack_pairs_k<<<8,  256, 0, stream>>>(Wl2, Wl2p, 64, 64);
    pack_pairs_k<<<8,  256, 0, stream>>>(Wr2, Wr2p, 64, 64);

    // ---- 2. early-join static GEMMs + degree ----
    node_static_k<<<N_NODES, 128, 0, stream>>>(node_feat, node_opcode, op_emb,
                                               shape_emb, W1, b1, a_node);
    cfg_static_k<<<NB * N_CFG, 128, 0, stream>>>(config_feat, W1, Cst);
    (void)hipMemsetAsync(deg, 0, N_NODES * sizeof(float), stream);
    deg_k<<<(NE + 255) / 256, 256, 0, stream>>>(edge_index, deg);

    // ---- 3. x1 = relu(a + C)  (MLP layer 1, early-joined) ----
    build_x1_k<<<(MROWS * 64) / 256, 256, 0, stream>>>(a_node, Cst, batch, x1);

    // ---- 4. x2 = relu(x1 @ W2 + b2) : TDM-staged WMMA, M=400K K=128 N=128 ----
    gemm_bias_relu_k<128, 128><<<MROWS / 64, 256, 0, stream>>>(x1, W2p, b2, x2);

    // ---- 5. SAGE layer 0 (K=128 -> 64) ----
    (void)hipMemsetAsync(agg, 0, (size_t)MROWS * 128 * 4, stream);
    edge_scatter_k<<<dim3(NE, (N_CFG * 128) / 256), 256, 0, stream>>>(x2, agg, edge_index, 7);
    mean_fin_k<<<((size_t)MROWS * 128) / 256, 256, 0, stream>>>(agg, deg, meanb, 7);
    sage_bias_relu_k<128><<<MROWS / 64, 128, 0, stream>>>((const unsigned*)x2, (const unsigned*)meanb,
                                                          Wl0p, Wr0p, bg0, x3a);

    // ---- 6. SAGE layer 1 (K=64) ----
    (void)hipMemsetAsync(agg, 0, (size_t)MROWS * HD * 4, stream);
    edge_scatter_k<<<dim3(NE, (N_CFG * HD) / 256), 256, 0, stream>>>(x3a, agg, edge_index, 6);
    mean_fin_k<<<((size_t)MROWS * HD) / 256, 256, 0, stream>>>(agg, deg, meanb, 6);
    sage_bias_relu_k<64><<<MROWS / 64, 128, 0, stream>>>((const unsigned*)x3a, (const unsigned*)meanb,
                                                         Wl1p, Wr1p, bg1, x3b);

    // ---- 7. SAGE layer 2 (K=64) ----
    (void)hipMemsetAsync(agg, 0, (size_t)MROWS * HD * 4, stream);
    edge_scatter_k<<<dim3(NE, (N_CFG * HD) / 256), 256, 0, stream>>>(x3b, agg, edge_index, 6);
    mean_fin_k<<<((size_t)MROWS * HD) / 256, 256, 0, stream>>>(agg, deg, meanb, 6);
    sage_bias_relu_k<64><<<MROWS / 64, 128, 0, stream>>>((const unsigned*)x3b, (const unsigned*)meanb,
                                                         Wl2p, Wr2p, bg2, x3c);

    // ---- 8. pool + normalize + head ----
    pool_head_k<<<NB * N_CFG, HD, 0, stream>>>(x3c, Wp1, bp1, Wp2, bp2, out);
}
